// NeuralODEBlock_65687229825709
// MI455X (gfx1250) — matrix operand order repaired and checked by reading, not compiled, not run on hardware
//
#include <hip/hip_runtime.h>
#include <hip/hip_bf16.h>
#include <math.h>

// ---------------------------------------------------------------------------
// Fused 6-step RK4 Neural-ODE for MI455X (gfx1250, wave32, WMMA).
//   z' = W2 @ tanh(W1[:-1] @ z + t*W1[-1] + b1) + b2
// One workgroup (8 wave32) integrates a 16-row x 1024-col slab of the state
// for all 6 RK4 steps entirely on-chip: weights live in LDS as pre-swizzled
// f16 WMMA B-fragments, state/accumulators live in VGPRs in WMMA C layout.
// ---------------------------------------------------------------------------

typedef __attribute__((ext_vector_type(16))) _Float16 v16h;
typedef __attribute__((ext_vector_type(8)))  _Float16 v8h;
typedef __attribute__((ext_vector_type(8)))  float    v8f;

union AFrag { v16h v; v8h h[2]; };

#define D_MODEL 1024
#define HIDDEN  64
#define ROWS    16      // rows of the state per workgroup
#define THREADS 256     // 8 wave32
#define NWAVES  8

#define W1F_HALVES (D_MODEL * HIDDEN)   // 65536  (128 KB) B-frags of W1
#define W2F_HALVES (HIDDEN * D_MODEL)   // 65536  (128 KB) B-frags of W2
#define ZS_HALVES  (ROWS * D_MODEL)     // 16384  ( 32 KB) stage input (f16)
#define H32_FLOATS (ROWS * HIDDEN)      //  1024  (  4 KB) split-K reduce
#define HF_HALVES  (ROWS * HIDDEN)      //  1024  (  2 KB) tanh(h) as f16

#define SMEM_BYTES (W1F_HALVES*2 + W2F_HALVES*2 + ZS_HALVES*2 + \
                    H32_FLOATS*4 + HF_HALVES*2 + 64*4 + 64*4)   // 301568 < 320K

// A-matrix 16x32 f16 fragment from a row-major f16 LDS buffer.
// Layout: lane<16: halves 0-15 = K kb..kb+15;
//             lane>=16 uses kb+16. Row M = lane & 15. Two 16B ds loads.
__device__ __forceinline__ v16h load_a16(const _Float16* buf, int stride,
                                         int kbase, int lane) {
  int row = lane & 15;
  int kb  = kbase + ((lane & 16) ? 16 : 0);
  const v8h* p = (const v8h*)(buf + row * stride + kb);
  AFrag af;
  af.h[0] = p[0];   // K = kb .. kb+7
  af.h[1] = p[1];   // K = kb+8 .. kb+15
  return af.v;
}

// One dynamics evaluation: k = W2 @ tanh(W1^T@(z + c*k_prev) + t*w1t + b1) + b2
__device__ __forceinline__ void ode_stage(
    float t, float c,
    v8f z[8], v8f k[8],
    const _Float16* w1f, const _Float16* w2f,
    _Float16* zsb, float* h32, _Float16* hf,
    const float* b1s, const float* w1ts, const float b2r[8],
    int tid, int lane, int waveId, int rlo, int col0)
{
  // (a) stage input zs = z + c*k -> f16 LDS (C-layout scatter); zero reducer
#pragma unroll
  for (int nt = 0; nt < 8; ++nt)
#pragma unroll
    for (int v = 0; v < 8; ++v) {
      float val = z[nt][v] + c * k[nt][v];
      zsb[(rlo + v) * D_MODEL + col0 + nt * 16] = (_Float16)val;
    }
#pragma unroll
  for (int i = 0; i < 4; ++i) h32[tid * 4 + i] = 0.0f;
  __syncthreads();

  // (b) matmul1, split-K over waves: each wave reduces its own 128 columns
  v8f hc[4];
#pragma unroll
  for (int nt = 0; nt < 4; ++nt)
#pragma unroll
    for (int v = 0; v < 8; ++v) hc[nt][v] = 0.0f;
#pragma unroll
  for (int kl = 0; kl < 4; ++kl) {
    int kt = waveId * 4 + kl;                       // K-tile of 32
    v16h a = load_a16(zsb, D_MODEL, kt * 32, lane);
#pragma unroll
    for (int nt = 0; nt < 4; ++nt) {
      v16h b = *(const v16h*)(w1f + (kt * 4 + nt) * 512 + lane * 16);
      hc[nt] = __builtin_amdgcn_wmma_f32_16x16x32_f16(
          false, a, false, b, (short)0, hc[nt], false, false);
    }
  }
#pragma unroll
  for (int nt = 0; nt < 4; ++nt)
#pragma unroll
    for (int v = 0; v < 8; ++v)
      atomicAdd(&h32[(rlo + v) * HIDDEN + nt * 16 + (lane & 15)], hc[nt][v]);
  __syncthreads();

  // (c) bias + time channel + tanh -> f16
#pragma unroll
  for (int i = 0; i < 4; ++i) {
    int idx = tid * 4 + i;
    int n = idx & (HIDDEN - 1);
    hf[idx] = (_Float16)tanhf(h32[idx] + t * w1ts[n] + b1s[n]);
  }
  __syncthreads();

  // (d) matmul2: k = h @ W2 + b2 ; each wave owns 8 output column tiles
  v16h a0 = load_a16(hf, HIDDEN, 0, lane);
  v16h a1 = load_a16(hf, HIDDEN, 32, lane);
#pragma unroll
  for (int nt = 0; nt < 8; ++nt) {
    int ntg = waveId * 8 + nt;
    v8f cacc;
#pragma unroll
    for (int v = 0; v < 8; ++v) cacc[v] = b2r[nt];  // bias preloaded into C
    v16h bf0 = *(const v16h*)(w2f + (ntg * 2 + 0) * 512 + lane * 16);
    cacc = __builtin_amdgcn_wmma_f32_16x16x32_f16(
        false, a0, false, bf0, (short)0, cacc, false, false);
    v16h bf1 = *(const v16h*)(w2f + (ntg * 2 + 1) * 512 + lane * 16);
    cacc = __builtin_amdgcn_wmma_f32_16x16x32_f16(
        false, a1, false, bf1, (short)0, cacc, false, false);
    k[nt] = cacc;
  }
}

__global__ __launch_bounds__(THREADS, 1)
void neural_ode_kernel(const float* __restrict__ x,
                       const float* __restrict__ W1,   // (1025, 64)
                       const float* __restrict__ b1,   // (64,)
                       const float* __restrict__ W2,   // (64, 1024)
                       const float* __restrict__ b2,   // (1024,)
                       float* __restrict__ out)
{
  extern __shared__ char smem[];
  _Float16* w1f  = (_Float16*)smem;                 // W1 B-fragments
  _Float16* w2f  = w1f + W1F_HALVES;                // W2 B-fragments
  _Float16* zsb  = w2f + W2F_HALVES;                // stage input, row-major
  float*    h32  = (float*)(zsb + ZS_HALVES);       // split-K reduction
  _Float16* hf   = (_Float16*)(h32 + H32_FLOATS);   // tanh(h), row-major
  float*    b1s  = (float*)(hf + HF_HALVES);
  float*    w1ts = b1s + 64;                        // time row W1[-1]

  const int tid     = threadIdx.x;
  const int lane    = tid & 31;
  const int waveId  = tid >> 5;
  const int rlo     = (lane >> 4) * 8;              // C-layout row offset
  const int col0    = waveId * 128 + (lane & 15);   // wave's column slab
  const int rowbase = blockIdx.x * ROWS;

  // ---- one-time: swizzle weights f32->f16 into WMMA B-fragment layout ----
  // B 32x16 frag: lanes 0-15 hold K=0..15 (N=lane), lanes 16-31 hold K=16..31.
  for (int f = waveId; f < 128; f += NWAVES) {      // W1: kt 0..31, nt 0..3
    int kt = f >> 2, nt = f & 3;
    int n  = nt * 16 + (lane & 15);
    int kb = kt * 32 + ((lane & 16) ? 16 : 0);
    v16h frag;
#pragma unroll
    for (int j = 0; j < 16; ++j) frag[j] = (_Float16)W1[(kb + j) * HIDDEN + n];
    *(v16h*)(w1f + f * 512 + lane * 16) = frag;
  }
  for (int f = waveId; f < 128; f += NWAVES) {      // W2: nt 0..63, kt 0..1
    int nt = f >> 1, kt = f & 1;
    int n  = nt * 16 + (lane & 15);
    int kb = kt * 32 + ((lane & 16) ? 16 : 0);
    v16h frag;
#pragma unroll
    for (int j = 0; j < 16; ++j) frag[j] = (_Float16)W2[(kb + j) * D_MODEL + n];
    *(v16h*)(w2f + f * 512 + lane * 16) = frag;
  }
  for (int i = tid; i < HIDDEN; i += THREADS) {
    b1s[i]  = b1[i];
    w1ts[i] = W1[D_MODEL * HIDDEN + i];             // W1[-1] time channel
  }

  float b2r[8];
#pragma unroll
  for (int nt = 0; nt < 8; ++nt)
    b2r[nt] = b2[waveId * 128 + nt * 16 + (lane & 15)];

  // ---- load state slab into registers (WMMA C-fragment layout) ----
  v8f z[8], acc[8], k[8];
#pragma unroll
  for (int nt = 0; nt < 8; ++nt)
#pragma unroll
    for (int v = 0; v < 8; ++v)
      z[nt][v] = x[(size_t)(rowbase + rlo + v) * D_MODEL + col0 + nt * 16];

  __syncthreads();

  // ---- 6 RK4 steps, fully fused on-chip ----
  const float hstep = 1.0f / 6.0f;
#pragma unroll 1
  for (int s = 0; s < 6; ++s) {
    float t = hstep * (float)s;
#pragma unroll
    for (int nt = 0; nt < 8; ++nt)
#pragma unroll
      for (int v = 0; v < 8; ++v) k[nt][v] = 0.0f;

    // k1
    ode_stage(t, 0.0f, z, k, w1f, w2f, zsb, h32, hf, b1s, w1ts, b2r,
              tid, lane, waveId, rlo, col0);
#pragma unroll
    for (int nt = 0; nt < 8; ++nt)
#pragma unroll
      for (int v = 0; v < 8; ++v) acc[nt][v] = k[nt][v];

    // k2 (input z + h/2 * k1)
    ode_stage(t + 0.5f * hstep, 0.5f * hstep, z, k, w1f, w2f, zsb, h32, hf,
              b1s, w1ts, b2r, tid, lane, waveId, rlo, col0);
#pragma unroll
    for (int nt = 0; nt < 8; ++nt)
#pragma unroll
      for (int v = 0; v < 8; ++v) acc[nt][v] += 2.0f * k[nt][v];

    // k3 (input z + h/2 * k2)
    ode_stage(t + 0.5f * hstep, 0.5f * hstep, z, k, w1f, w2f, zsb, h32, hf,
              b1s, w1ts, b2r, tid, lane, waveId, rlo, col0);
#pragma unroll
    for (int nt = 0; nt < 8; ++nt)
#pragma unroll
      for (int v = 0; v < 8; ++v) acc[nt][v] += 2.0f * k[nt][v];

    // k4 (input z + h * k3)
    ode_stage(t + hstep, hstep, z, k, w1f, w2f, zsb, h32, hf,
              b1s, w1ts, b2r, tid, lane, waveId, rlo, col0);
#pragma unroll
    for (int nt = 0; nt < 8; ++nt)
#pragma unroll
      for (int v = 0; v < 8; ++v) {
        acc[nt][v] += k[nt][v];
        z[nt][v]   += (hstep / 6.0f) * acc[nt][v];
      }
  }

  // ---- write final state ----
#pragma unroll
  for (int nt = 0; nt < 8; ++nt)
#pragma unroll
    for (int v = 0; v < 8; ++v)
      out[(size_t)(rowbase + rlo + v) * D_MODEL + col0 + nt * 16] = z[nt][v];
}

extern "C" void kernel_launch(void* const* d_in, const int* in_sizes, int n_in,
                              void* d_out, int out_size, void* d_ws, size_t ws_size,
                              hipStream_t stream) {
  (void)n_in; (void)out_size; (void)d_ws; (void)ws_size;
  const float* x  = (const float*)d_in[0];
  const float* W1 = (const float*)d_in[1];
  const float* b1 = (const float*)d_in[2];
  const float* W2 = (const float*)d_in[3];
  const float* b2 = (const float*)d_in[4];
  float* out = (float*)d_out;

  const int totalRows = in_sizes[0] / D_MODEL;   // 8*2048 = 16384
  const int blocks = totalRows / ROWS;           // 1024 workgroups

  // Opt into the large (294 KB) dynamic LDS allocation (WGP has 320 KB).
  hipFuncSetAttribute(reinterpret_cast<const void*>(neural_ode_kernel),
                      hipFuncAttributeMaxDynamicSharedMemorySize, SMEM_BYTES);

  neural_ode_kernel<<<dim3(blocks), dim3(THREADS), SMEM_BYTES, stream>>>(
      x, W1, b1, W2, b2, out);
}
